// SplitHeadSAMEncoderAttention_8564164788736
// MI455X (gfx1250) — compile-verified, matrix operand order checked by reading, not run on hardware
//
#include <hip/hip_runtime.h>
#include <stdint.h>

// ---------------------------------------------------------------------------
// CDNA5 (gfx1250) wave32 WMMA types and helpers
// ---------------------------------------------------------------------------
typedef __attribute__((ext_vector_type(16))) __bf16 v16bf;
typedef __attribute__((ext_vector_type(8)))  __bf16 v8bf;
typedef __attribute__((ext_vector_type(8)))  float  v8f;

union FragBF { v16bf v; v8bf h[2]; };

__device__ __forceinline__ unsigned short f2bf(float f) {
  unsigned int u = __builtin_bit_cast(unsigned int, f);
  u += 0x7FFFu + ((u >> 16) & 1u);          // round-to-nearest-even
  return (unsigned short)(u >> 16);
}
__device__ __forceinline__ float bf2f(unsigned short h) {
  unsigned int u = ((unsigned int)h) << 16;
  return __builtin_bit_cast(float, u);
}

// Load a 16x32 bf16 A/B fragment for one lane from LDS.
// Lane-group g (lane>>4) holds K = {lo..lo+7} in halves [0..7] and
// K = {hi..hi+7} in halves [8..15]  (lo = kk + 8g, hi = kk + 16 + 8g).
__device__ __forceinline__ v16bf ld_frag(const unsigned short* p, int lo, int hi) {
  FragBF f;
  f.h[0] = *(const v8bf*)(p + lo);
  f.h[1] = *(const v8bf*)(p + hi);
  return f.v;
}

#define WMMA_BF16(a, b, c) \
  __builtin_amdgcn_wmma_f32_16x16x32_bf16(false, (a), false, (b), (short)0, (c), false, false)

// CDNA5 async global->LDS copy (16B per lane), tracked by ASYNCcnt.
__device__ __forceinline__ void async_ld_b128(void* lds_dst, const void* gsrc) {
  unsigned int loff = (unsigned int)(uintptr_t)lds_dst;
  asm volatile("global_load_async_to_lds_b128 %0, %1, off"
               :: "v"(loff), "v"(gsrc) : "memory");
}
__device__ __forceinline__ void async_wait0() {
  asm volatile("s_wait_asynccnt 0x0" ::: "memory");
}

// Problem constants
#define NB    4     // batch
#define NHEAD 12
#define DHEAD 64
#define SSEQ  1024  // 32*32
#define CDIM  768

// ---------------------------------------------------------------------------
// Kernel 1: QKV projection GEMM  (M=4096, N=768, K=768) in bf16 WMMA.
// blockIdx.z selects q/k/v.  Block = 256 thr (8 waves), tile 256x64, BK=64;
// each wave owns 32 output rows (2 A-fragments reuse the 4 B-fragments).
// q,k stored (b,n,s,d) bf16; v stored transposed (b,n,d,s) bf16.
// ---------------------------------------------------------------------------
__global__ __launch_bounds__(256) void qkv_gemm_kernel(
    const float* __restrict__ x,
    const float* __restrict__ Wq, const float* __restrict__ Wk, const float* __restrict__ Wv,
    const float* __restrict__ bq, const float* __restrict__ bk, const float* __restrict__ bv,
    unsigned short* __restrict__ qo, unsigned short* __restrict__ ko,
    unsigned short* __restrict__ vTo)
{
  constexpr int LDT = 72;                       // 144B row stride: 16B aligned, conflict-free
  __shared__ unsigned short shA[256 * LDT];
  __shared__ unsigned short shB[64 * LDT];

  const int which = blockIdx.z;
  const float* W    = (which == 0) ? Wq : (which == 1) ? Wk : Wv;
  const float* bias = (which == 0) ? bq : (which == 1) ? bk : bv;

  const int tid  = threadIdx.x;
  const int wave = tid >> 5;
  const int lane = tid & 31;
  const int g    = lane >> 4;
  const int ln   = lane & 15;

  const int m0   = blockIdx.x * 256;
  const int head = blockIdx.y;                  // BN == d == 64 -> n-tile == head
  const int n0   = head * 64;

  v8f zero = {0.f,0.f,0.f,0.f,0.f,0.f,0.f,0.f};
  v8f acc[2][4] = {{zero, zero, zero, zero}, {zero, zero, zero, zero}};

  const int brow = tid >> 2, bpart = tid & 3;   // B: 4 threads/row, 16 floats each

  for (int kb = 0; kb < CDIM; kb += 64) {
    __syncthreads();
    { // stage A tile 256x64 (x, fp32 -> bf16): one full row per thread
      const float4* src = (const float4*)(x + (size_t)(m0 + tid) * CDIM + kb);
      unsigned short* dst = &shA[tid * LDT];
      #pragma unroll
      for (int i = 0; i < 16; ++i) {
        float4 f = src[i];
        *(ushort4*)(dst + i * 4) = make_ushort4(f2bf(f.x), f2bf(f.y), f2bf(f.z), f2bf(f.w));
      }
    }
    { // stage B tile 64x64 (W row-major over K, fp32 -> bf16)
      const float4* src = (const float4*)(W + (size_t)(n0 + brow) * CDIM + kb + bpart * 16);
      unsigned short* dst = &shB[brow * LDT + bpart * 16];
      #pragma unroll
      for (int i = 0; i < 4; ++i) {
        float4 f = src[i];
        *(ushort4*)(dst + i * 4) = make_ushort4(f2bf(f.x), f2bf(f.y), f2bf(f.z), f2bf(f.w));
      }
    }
    __syncthreads();

    #pragma unroll
    for (int kk = 0; kk < 64; kk += 32) {
      v16bf a0 = ld_frag(&shA[(wave * 32 + ln) * LDT],      kk + 8 * g, kk + 16 + 8 * g);
      v16bf a1 = ld_frag(&shA[(wave * 32 + 16 + ln) * LDT], kk + 8 * g, kk + 16 + 8 * g);
      v16bf bf[4];
      #pragma unroll
      for (int nt = 0; nt < 4; ++nt)
        bf[nt] = ld_frag(&shB[(nt * 16 + ln) * LDT], kk + 8 * g, kk + 16 + 8 * g);
      #pragma unroll
      for (int nt = 0; nt < 4; ++nt) {
        acc[0][nt] = WMMA_BF16(a0, bf[nt], acc[0][nt]);
        acc[1][nt] = WMMA_BF16(a1, bf[nt], acc[1][nt]);
      }
    }
  }

  // epilogue: + bias, store bf16 (v transposed)
  #pragma unroll
  for (int mt = 0; mt < 2; ++mt) {
    #pragma unroll
    for (int nt = 0; nt < 4; ++nt) {
      int dd = nt * 16 + ln;
      float bv_ = bias[head * DHEAD + dd];
      #pragma unroll
      for (int r = 0; r < 8; ++r) {
        int M  = m0 + wave * 32 + mt * 16 + r + 8 * g;   // C-layout: row = r + 8g
        int bb = M >> 10, s = M & 1023;
        unsigned short hval = f2bf(acc[mt][nt][r] + bv_);
        if (which == 0)
          qo[((size_t)(bb * NHEAD + head) * SSEQ + s) * DHEAD + dd] = hval;
        else if (which == 1)
          ko[((size_t)(bb * NHEAD + head) * SSEQ + s) * DHEAD + dd] = hval;
        else
          vTo[((size_t)(bb * NHEAD + head) * DHEAD + dd) * SSEQ + s] = hval;
      }
    }
  }
}

// ---------------------------------------------------------------------------
// Kernel 2: decomposed relative-position bias tables.
// rel_h[bn,s,kh] = q[bn,s,:] . rel_pos_h[qh-kh+31,:]   (and same for w).
// One thread per (bn, s, j), j<32 -> kh, j>=32 -> kw.  ~0.4 GFLOP.
// ---------------------------------------------------------------------------
__global__ __launch_bounds__(256) void relbias_kernel(
    const unsigned short* __restrict__ qb,
    const float* __restrict__ rph, const float* __restrict__ rpw,
    float* __restrict__ relh, float* __restrict__ relw)
{
  int idx = blockIdx.x * 256 + threadIdx.x;
  int j  = idx & 63;
  int s  = (idx >> 6) & 1023;
  int bn = idx >> 16;
  const unsigned short* qrow = qb + ((size_t)bn * SSEQ + s) * DHEAD;
  const float* table;
  if (j < 32) table = rph + (size_t)((s >> 5) - j + 31) * DHEAD;
  else        table = rpw + (size_t)((s & 31) - (j - 32) + 31) * DHEAD;
  float acc = 0.f;
  #pragma unroll 8
  for (int d = 0; d < DHEAD; ++d) acc += bf2f(qrow[d]) * table[d];
  if (j < 32) relh[((size_t)bn * SSEQ + s) * 32 + j]        = acc;
  else        relw[((size_t)bn * SSEQ + s) * 32 + (j - 32)] = acc;
}

// ---------------------------------------------------------------------------
// Kernel 3: fused flash attention per (head, 64-row q tile).
// Block = 128 thr (4 waves); wave w owns q rows [w*16, w*16+16).
// scores/PV both WMMA bf16; online softmax in registers; P re-fragmented
// through LDS.  K/V/Q tiles staged with GLOBAL_LOAD_ASYNC_TO_LDS_B128.
// ---------------------------------------------------------------------------
__global__ __launch_bounds__(128) void flash_attn_kernel(
    const unsigned short* __restrict__ qb,
    const unsigned short* __restrict__ kb,
    const unsigned short* __restrict__ vT,
    const float* __restrict__ relh, const float* __restrict__ relw,
    unsigned short* __restrict__ attnA)
{
  constexpr int LDT = 72;                        // 144B row stride, 16B aligned
  __shared__ unsigned short shQ[64 * LDT];
  __shared__ unsigned short shK[64 * LDT];
  __shared__ unsigned short shV[64 * LDT];
  __shared__ unsigned short shP[64 * LDT];
  __shared__ float shRh[64 * 32];
  __shared__ float shRw[64 * 32];

  const int qt   = blockIdx.x;                   // q tile (0..15)
  const int bn   = blockIdx.y;                   // b*12 + head (0..47)
  const int tid  = threadIdx.x;
  const int wave = tid >> 5;
  const int lane = tid & 31;
  const int g    = lane >> 4;
  const int ln   = lane & 15;
  const float scale = 0.125f;                    // d^-0.5, d = 64

  const int lrow  = tid >> 1;                    // 0..63
  const int lpart = tid & 1;                     // 32-halves half-row

  { // stage Q tile (64x64 bf16) once, async
    const unsigned short* src = qb + ((size_t)bn * SSEQ + qt * 64 + lrow) * DHEAD + lpart * 32;
    unsigned short* dst = &shQ[lrow * LDT + lpart * 32];
    #pragma unroll
    for (int i = 0; i < 4; ++i) async_ld_b128(dst + i * 8, src + i * 8);
  }
  { // stage rel bias rows (cover every kh/kw for these 64 q rows)
    const float4* sh_ = (const float4*)(relh + ((size_t)bn * SSEQ + qt * 64 + lrow) * 32) + lpart * 4;
    const float4* sw_ = (const float4*)(relw + ((size_t)bn * SSEQ + qt * 64 + lrow) * 32) + lpart * 4;
    float4* dh = (float4*)(&shRh[lrow * 32]) + lpart * 4;
    float4* dw = (float4*)(&shRw[lrow * 32]) + lpart * 4;
    #pragma unroll
    for (int i = 0; i < 4; ++i) { dh[i] = sh_[i]; dw[i] = sw_[i]; }
  }

  float mi[8], li[8];
  v8f zero = {0.f,0.f,0.f,0.f,0.f,0.f,0.f,0.f};
  v8f o[4] = {zero, zero, zero, zero};
  #pragma unroll
  for (int r = 0; r < 8; ++r) { mi[r] = -1e30f; li[r] = 0.f; }

  for (int kt = 0; kt < 16; ++kt) {
    __syncthreads();                             // protect shK/shV vs previous PV phase
    { // stage K tile (64 k-rows x 64 d), async
      const unsigned short* src = kb + ((size_t)bn * SSEQ + kt * 64 + lrow) * DHEAD + lpart * 32;
      unsigned short* dst = &shK[lrow * LDT + lpart * 32];
      #pragma unroll
      for (int i = 0; i < 4; ++i) async_ld_b128(dst + i * 8, src + i * 8);
    }
    { // stage V^T tile (64 d-rows x 64 k), async
      const unsigned short* src = vT + ((size_t)bn * DHEAD + lrow) * SSEQ + kt * 64 + lpart * 32;
      unsigned short* dst = &shV[lrow * LDT + lpart * 32];
      #pragma unroll
      for (int i = 0; i < 4; ++i) async_ld_b128(dst + i * 8, src + i * 8);
    }
    if (kt + 1 < 16)                             // global_prefetch_b8 for next K tile
      __builtin_prefetch(kb + ((size_t)bn * SSEQ + (kt + 1) * 64 + lrow) * DHEAD, 0, 1);
    async_wait0();                               // ASYNCcnt -> 0 before the barrier
    __syncthreads();

    // ---- scores = Q @ K^T : 16x64 per wave, K-dim = d = 64 (2x4 wmma) ----
    v8f sc[4] = {zero, zero, zero, zero};
    #pragma unroll
    for (int kk = 0; kk < 64; kk += 32) {
      v16bf a = ld_frag(&shQ[(wave * 16 + ln) * LDT], kk + 8 * g, kk + 16 + 8 * g);
      v16bf bf[4];
      #pragma unroll
      for (int nt = 0; nt < 4; ++nt)
        bf[nt] = ld_frag(&shK[(nt * 16 + ln) * LDT], kk + 8 * g, kk + 16 + 8 * g);
      #pragma unroll
      for (int nt = 0; nt < 4; ++nt)
        sc[nt] = WMMA_BF16(a, bf[nt], sc[nt]);
    }

    // ---- scale + rel bias + online softmax ----
    float tmax[8];
    #pragma unroll
    for (int r = 0; r < 8; ++r) tmax[r] = -1e30f;
    #pragma unroll
    for (int nt = 0; nt < 4; ++nt) {
      int nl = nt * 16 + ln;                     // k index within tile
      int kh = kt * 2 + (nl >> 5);
      int kw = nl & 31;
      #pragma unroll
      for (int r = 0; r < 8; ++r) {
        int qr = wave * 16 + r + 8 * g;          // local q row
        float sv = sc[nt][r] * scale + shRh[qr * 32 + kh] + shRw[qr * 32 + kw];
        sc[nt][r] = sv;
        tmax[r] = fmaxf(tmax[r], sv);
      }
    }
    #pragma unroll
    for (int off = 8; off >= 1; off >>= 1) {     // reduce over the 16-lane N group
      #pragma unroll
      for (int r = 0; r < 8; ++r)
        tmax[r] = fmaxf(tmax[r], __shfl_xor(tmax[r], off, 32));
    }

    float alpha[8], tsum[8];
    #pragma unroll
    for (int r = 0; r < 8; ++r) {
      float nm = fmaxf(mi[r], tmax[r]);
      alpha[r] = __expf(mi[r] - nm);
      mi[r] = nm;
      tsum[r] = 0.f;
    }
    #pragma unroll
    for (int nt = 0; nt < 4; ++nt) {
      #pragma unroll
      for (int r = 0; r < 8; ++r) {
        float p = __expf(sc[nt][r] - mi[r]);
        sc[nt][r] = p;
        tsum[r] += p;
      }
    }
    #pragma unroll
    for (int off = 8; off >= 1; off >>= 1) {
      #pragma unroll
      for (int r = 0; r < 8; ++r)
        tsum[r] += __shfl_xor(tsum[r], off, 32);
    }
    #pragma unroll
    for (int r = 0; r < 8; ++r) li[r] = li[r] * alpha[r] + tsum[r];
    #pragma unroll
    for (int nt = 0; nt < 4; ++nt) {
      #pragma unroll
      for (int r = 0; r < 8; ++r) o[nt][r] *= alpha[r];
    }

    // ---- re-fragment P through LDS (C layout -> A layout) ----
    #pragma unroll
    for (int nt = 0; nt < 4; ++nt) {
      #pragma unroll
      for (int r = 0; r < 8; ++r)
        shP[(wave * 16 + r + 8 * g) * LDT + nt * 16 + ln] = f2bf(sc[nt][r]);
    }
    __syncthreads();

    // ---- O += P @ V : K-dim = 64 k-cols (2x4 wmma) ----
    #pragma unroll
    for (int kk = 0; kk < 64; kk += 32) {
      v16bf a = ld_frag(&shP[(wave * 16 + ln) * LDT], kk + 8 * g, kk + 16 + 8 * g);
      v16bf bf[4];
      #pragma unroll
      for (int nt = 0; nt < 4; ++nt)
        bf[nt] = ld_frag(&shV[(nt * 16 + ln) * LDT], kk + 8 * g, kk + 16 + 8 * g);
      #pragma unroll
      for (int nt = 0; nt < 4; ++nt)
        o[nt] = WMMA_BF16(a, bf[nt], o[nt]);
    }
  }

  // normalize and store concat-head layout (b, s, head*64+dd) as bf16
  const int head = bn % NHEAD;
  const int bb   = bn / NHEAD;
  #pragma unroll
  for (int r = 0; r < 8; ++r) {
    float inv = 1.f / li[r];
    int srow = qt * 64 + wave * 16 + r + 8 * g;
    #pragma unroll
    for (int nt = 0; nt < 4; ++nt) {
      int col = head * DHEAD + nt * 16 + ln;
      attnA[((size_t)(bb * SSEQ + srow)) * CDIM + col] = f2bf(o[nt][r] * inv);
    }
  }
}

// ---------------------------------------------------------------------------
// Kernel 4: output projection GEMM (M=4096, N=768, K=768), fp32 out + bias.
// Tile 256x64, 2 M-subtiles per wave; A tile (already bf16) staged with
// async global->LDS copies.
// ---------------------------------------------------------------------------
__global__ __launch_bounds__(256) void proj_gemm_kernel(
    const unsigned short* __restrict__ attnA,
    const float* __restrict__ pw, const float* __restrict__ pb,
    float* __restrict__ out)
{
  constexpr int LDT = 72;
  __shared__ unsigned short shA[256 * LDT];
  __shared__ unsigned short shB[64 * LDT];

  const int tid  = threadIdx.x;
  const int wave = tid >> 5;
  const int lane = tid & 31;
  const int g    = lane >> 4;
  const int ln   = lane & 15;
  const int m0   = blockIdx.x * 256;
  const int n0   = blockIdx.y * 64;

  v8f zero = {0.f,0.f,0.f,0.f,0.f,0.f,0.f,0.f};
  v8f acc[2][4] = {{zero, zero, zero, zero}, {zero, zero, zero, zero}};

  const int brow = tid >> 2, bpart = tid & 3;   // 16 floats per thread

  for (int kb = 0; kb < CDIM; kb += 64) {
    __syncthreads();
    { // stage A tile 256x64 (already bf16): async, one row per thread
      const unsigned short* src = attnA + (size_t)(m0 + tid) * CDIM + kb;
      unsigned short* dst = &shA[tid * LDT];
      #pragma unroll
      for (int i = 0; i < 8; ++i) async_ld_b128(dst + i * 8, src + i * 8);
    }
    { // stage B tile 64x64 (proj_w fp32 -> bf16); out = A @ W^T
      const float4* src = (const float4*)(pw + (size_t)(n0 + brow) * CDIM + kb + bpart * 16);
      unsigned short* dst = &shB[brow * LDT + bpart * 16];
      #pragma unroll
      for (int i = 0; i < 4; ++i) {
        float4 f = src[i];
        *(ushort4*)(dst + i * 4) = make_ushort4(f2bf(f.x), f2bf(f.y), f2bf(f.z), f2bf(f.w));
      }
    }
    async_wait0();
    __syncthreads();

    #pragma unroll
    for (int kk = 0; kk < 64; kk += 32) {
      v16bf a0 = ld_frag(&shA[(wave * 32 + ln) * LDT],      kk + 8 * g, kk + 16 + 8 * g);
      v16bf a1 = ld_frag(&shA[(wave * 32 + 16 + ln) * LDT], kk + 8 * g, kk + 16 + 8 * g);
      v16bf bf[4];
      #pragma unroll
      for (int nt = 0; nt < 4; ++nt)
        bf[nt] = ld_frag(&shB[(nt * 16 + ln) * LDT], kk + 8 * g, kk + 16 + 8 * g);
      #pragma unroll
      for (int nt = 0; nt < 4; ++nt) {
        acc[0][nt] = WMMA_BF16(a0, bf[nt], acc[0][nt]);
        acc[1][nt] = WMMA_BF16(a1, bf[nt], acc[1][nt]);
      }
    }
  }

  #pragma unroll
  for (int mt = 0; mt < 2; ++mt) {
    #pragma unroll
    for (int nt = 0; nt < 4; ++nt) {
      int nc = n0 + nt * 16 + ln;
      float bias = pb[nc];
      #pragma unroll
      for (int r = 0; r < 8; ++r) {
        int M = m0 + wave * 32 + mt * 16 + r + 8 * g;
        out[(size_t)M * CDIM + nc] = acc[mt][nt][r] + bias;
      }
    }
  }
}

// ---------------------------------------------------------------------------
// Launch
// ---------------------------------------------------------------------------
extern "C" void kernel_launch(void* const* d_in, const int* in_sizes, int n_in,
                              void* d_out, int out_size, void* d_ws, size_t ws_size,
                              hipStream_t stream) {
  (void)in_sizes; (void)n_in; (void)out_size; (void)ws_size;
  const float* x   = (const float*)d_in[0];
  const float* Wq  = (const float*)d_in[1];
  const float* Wk  = (const float*)d_in[2];
  const float* Wv  = (const float*)d_in[3];
  const float* bq  = (const float*)d_in[4];
  const float* bk  = (const float*)d_in[5];
  const float* bv  = (const float*)d_in[6];
  const float* rph = (const float*)d_in[7];
  const float* rpw = (const float*)d_in[8];
  const float* pw  = (const float*)d_in[9];
  const float* pb  = (const float*)d_in[10];
  float* out = (float*)d_out;

  // workspace layout (bytes): q/k/vT bf16 (3 x 6 MB), rel_h/rel_w f32 (2 x 6 MB),
  // attn_out bf16 (6 MB)  -> 36 MB total
  char* ws = (char*)d_ws;
  unsigned short* qB    = (unsigned short*)(ws);
  unsigned short* kB    = (unsigned short*)(ws + 6291456);
  unsigned short* vTB   = (unsigned short*)(ws + 12582912);
  float*          relh  = (float*)(ws + 18874368);
  float*          relw  = (float*)(ws + 25165824);
  unsigned short* attnA = (unsigned short*)(ws + 31457280);

  qkv_gemm_kernel<<<dim3(16, 12, 3), 256, 0, stream>>>(x, Wq, Wk, Wv, bq, bk, bv, qB, kB, vTB);
  relbias_kernel<<<dim3(12288), 256, 0, stream>>>(qB, rph, rpw, relh, relw);
  flash_attn_kernel<<<dim3(16, 48), 128, 0, stream>>>(qB, kB, vTB, relh, relw, attnA);
  proj_gemm_kernel<<<dim3(16, 12), 256, 0, stream>>>(attnA, pw, pb, out);
}
